// MGDN_51110110822728
// MI455X (gfx1250) — compile-verified, matrix operand order checked by reading, not run on hardware
//
#include <hip/hip_runtime.h>
#include <hip/hip_bf16.h>

typedef __attribute__((ext_vector_type(16))) _Float16 v16h;
typedef __attribute__((ext_vector_type(8)))  _Float16 v8h;
typedef __attribute__((ext_vector_type(8)))  float    v8f;

#define TOPK 20
#define FDIM 64

// ---------------------------------------------------------------------------
// Row L2-normalize + pack to f16: wn16[n,:] = f16(emb[n,:] / ||emb[n,:]||),
// rows n in [N, Np) zero-filled so the GEMM needs no row bounds checks.
// ---------------------------------------------------------------------------
__global__ void normalize_pack_kernel(const float* __restrict__ emb,
                                      _Float16* __restrict__ wn16,
                                      int N, int Np) {
  int n = blockIdx.x * blockDim.x + threadIdx.x;
  if (n >= Np) return;
  if (n >= N) {
    #pragma unroll
    for (int f = 0; f < FDIM; ++f) wn16[(size_t)n * FDIM + f] = (_Float16)0.f;
    return;
  }
  float s = 0.f;
  #pragma unroll
  for (int f = 0; f < FDIM; ++f) { float v = emb[n * FDIM + f]; s += v * v; }
  float inv = rsqrtf(s);
  #pragma unroll
  for (int f = 0; f < FDIM; ++f)
    wn16[(size_t)n * FDIM + f] = (_Float16)(emb[n * FDIM + f] * inv);
}

// ---------------------------------------------------------------------------
// Pack f32 A[M x lda] -> f16 [Mp x Kp], zero-padded (k>=K or m>=M -> 0).
// ---------------------------------------------------------------------------
__global__ void pack_a_kernel(const float* __restrict__ src, int lda,
                              int M, int K, _Float16* __restrict__ dst,
                              int Kp, int total) {
  int tid = blockIdx.x * blockDim.x + threadIdx.x;
  if (tid >= total) return;
  int k = tid % Kp, m = tid / Kp;
  float v = (m < M && k < K) ? src[(size_t)m * lda + k] : 0.f;
  dst[tid] = (_Float16)v;
}

// ---------------------------------------------------------------------------
// Pack B transposed -> f16 [Ncp x Kp]: dst[n*Kp+k] = B[k][n].
// transB==0: src is [K x Nc] row-major (ldb = Nc). transB==1: src is [Nc x K].
// ---------------------------------------------------------------------------
__global__ void pack_bt_kernel(const float* __restrict__ src, int ldb, int transB,
                               int K, int Nc, _Float16* __restrict__ dst,
                               int Kp, int total) {
  int tid = blockIdx.x * blockDim.x + threadIdx.x;
  if (tid >= total) return;
  int k = tid % Kp, n = tid / Kp;
  float v = 0.f;
  if (n < Nc && k < K)
    v = transB ? src[(size_t)n * ldb + k] : src[(size_t)k * ldb + n];
  dst[tid] = (_Float16)v;
}

// ---------------------------------------------------------------------------
// Packed WMMA GEMM: C[M x Nc] = A16[Mp x Kp] * Bt16[Ncp x Kp]^T (+ bias).
// One wave per 16x16 tile; 4 waves/block over 4 N-tiles. Per K-chunk of 32:
// each lane loads two contiguous 16B groups from its A row and its Bt row
// (CDNA5 ISA 7.12.2 16-bit A/B striping), one v_wmma_f32_16x16x32_f16.
// Branch-free inner loop: EXEC all-ones through the WMMA.
// ---------------------------------------------------------------------------
__global__ void wmma_gemm_packed_kernel(const _Float16* __restrict__ A,
                                        const _Float16* __restrict__ Bt,
                                        const float* __restrict__ bias,
                                        float* __restrict__ C, int ldc,
                                        int M, int Kp, int Nc, int Ncp) {
  const int lane = threadIdx.x & 31;
  const int wave = threadIdx.x >> 5;
  const int half = lane >> 4;   // 0: lanes 0-15, 1: lanes 16-31
  const int l15  = lane & 15;

  const int tm = blockIdx.x;
  const int tn = blockIdx.y * 4 + wave;     // wave-uniform
  if (tn * 16 >= Ncp) return;               // wave-uniform skip

  const _Float16* ap = A  + (size_t)(tm * 16 + l15) * Kp + half * 8;
  const _Float16* bp = Bt + (size_t)(tn * 16 + l15) * Kp + half * 8;

  v8f acc = {};
  for (int kk = 0; kk < Kp; kk += 32) {
    v8h a0 = *(const v8h*)(ap + kk);        // K group {kk + half*8 .. +7}
    v8h a1 = *(const v8h*)(ap + kk + 16);   // K group {kk+16+half*8 .. +7}
    v8h b0 = *(const v8h*)(bp + kk);
    v8h b1 = *(const v8h*)(bp + kk + 16);
    v16h a = __builtin_shufflevector(a0, a1, 0,1,2,3,4,5,6,7,8,9,10,11,12,13,14,15);
    v16h b = __builtin_shufflevector(b0, b1, 0,1,2,3,4,5,6,7,8,9,10,11,12,13,14,15);
    acc = __builtin_amdgcn_wmma_f32_16x16x32_f16(
        /*neg_a=*/false, a, /*neg_b=*/false, b,
        /*c_mod=*/(short)0, acc, /*reuse_a=*/false, /*reuse_b=*/false);
  }

  // C/D f32 16x16 layout: VGPR i -> M = i + 8*half, N = l15.
  #pragma unroll
  for (int i = 0; i < 8; ++i) {
    int row = tm * 16 + (half << 3) + i;
    int col = tn * 16 + l15;
    if (row < M && col < Nc) {
      float v = acc[i];
      if (bias) v += bias[col];
      C[(size_t)row * ldc + col] = v;
    }
  }
}

// ---------------------------------------------------------------------------
// Top-20 per row, insertion select (ties: lower index first, like lax.top_k).
// ---------------------------------------------------------------------------
__global__ void topk20_kernel(const float* __restrict__ cos, int N,
                              int* __restrict__ idx_out,
                              float* __restrict__ fidx_out) {
  int row = blockIdx.x * blockDim.x + threadIdx.x;
  if (row >= N) return;
  float vals[TOPK]; int inds[TOPK];
  #pragma unroll
  for (int j = 0; j < TOPK; ++j) { vals[j] = -3.0e38f; inds[j] = 0; }
  const float* r = cos + (size_t)row * N;
  for (int c = 0; c < N; ++c) {
    float v = r[c];
    if (v > vals[TOPK - 1]) {
      int p = TOPK - 1;
      while (p > 0 && vals[p - 1] < v) {
        vals[p] = vals[p - 1]; inds[p] = inds[p - 1]; --p;
      }
      vals[p] = v; inds[p] = c;
    }
  }
  for (int j = 0; j < TOPK; ++j) {
    idx_out[row * TOPK + j] = inds[j];
    if (fidx_out) fidx_out[row * TOPK + j] = (float)inds[j];
  }
}

// ---------------------------------------------------------------------------
// GCN aggregate: out[b,n,f] = relu(mean_j h[b, idx[n,j], f]) (* emb[n,f])
// ---------------------------------------------------------------------------
__global__ void gather_mean_relu_kernel(const float* __restrict__ h,
                                        const int* __restrict__ idx,
                                        const float* __restrict__ emb,
                                        float* __restrict__ out,
                                        int N, int total) {
  int tid = blockIdx.x * blockDim.x + threadIdx.x;
  if (tid >= total) return;
  int f = tid & (FDIM - 1);
  int n = (tid / FDIM) % N;
  int b = tid / (FDIM * N);
  const int* nbr = idx + n * TOPK;
  float s = 0.f;
  #pragma unroll
  for (int j = 0; j < TOPK; ++j)
    s += h[((size_t)b * N + nbr[j]) * FDIM + f];
  s = fmaxf(s * (1.f / TOPK), 0.f);
  if (emb) s *= emb[n * FDIM + f];
  out[tid] = s;
}

// out[r,f] = a[r,f] * emb[r % N, f]
__global__ void rowmul_kernel(const float* __restrict__ a,
                              const float* __restrict__ emb,
                              float* __restrict__ out, int N, int total) {
  int tid = blockIdx.x * blockDim.x + threadIdx.x;
  if (tid >= total) return;
  int f = tid & (FDIM - 1);
  int n = (tid / FDIM) % N;
  out[tid] = a[tid] * emb[n * FDIM + f];
}

// Per-feature mean / rsqrt(var+eps) over R rows (population variance).
__global__ void colstats_kernel(const float* __restrict__ t, int R,
                                float* __restrict__ mean,
                                float* __restrict__ invstd) {
  __shared__ float ssum[256], ssq[256];
  int f = blockIdx.x;
  float s = 0.f, q = 0.f;
  for (int r = threadIdx.x; r < R; r += blockDim.x) {
    float v = t[(size_t)r * FDIM + f];
    s += v; q += v * v;
  }
  ssum[threadIdx.x] = s; ssq[threadIdx.x] = q;
  __syncthreads();
  for (int st = 128; st > 0; st >>= 1) {
    if (threadIdx.x < st) {
      ssum[threadIdx.x] += ssum[threadIdx.x + st];
      ssq[threadIdx.x]  += ssq[threadIdx.x + st];
    }
    __syncthreads();
  }
  if (threadIdx.x == 0) {
    float mu  = ssum[0] / (float)R;
    float var = ssq[0] / (float)R - mu * mu;
    mean[f]   = mu;
    invstd[f] = rsqrtf(var + 1e-5f);
  }
}

// out[r,f] = relu(g[f]*(t[r,f]-mu[f])*istd[f] + be[f])
__global__ void bn_apply_kernel(const float* __restrict__ t,
                                const float* __restrict__ g,
                                const float* __restrict__ be,
                                const float* __restrict__ mean,
                                const float* __restrict__ invstd,
                                float* __restrict__ out, int total) {
  int tid = blockIdx.x * blockDim.x + threadIdx.x;
  if (tid >= total) return;
  int f = tid & (FDIM - 1);
  float v = g[f] * (t[tid] - mean[f]) * invstd[f] + be[f];
  out[tid] = fmaxf(v, 0.f);
}

// out[r,o] = sum_f t[r,f]*W[f,o] + b[o]   (tiny NO = 1 or 3)
__global__ void proj_kernel(const float* __restrict__ t,
                            const float* __restrict__ W,
                            const float* __restrict__ b,
                            float* __restrict__ out, int R, int NO) {
  int tid = blockIdx.x * blockDim.x + threadIdx.x;
  if (tid >= R * NO) return;
  int o = tid % NO, r = tid / NO;
  float s = b[o];
  #pragma unroll
  for (int f = 0; f < FDIM; ++f) s += t[(size_t)r * FDIM + f] * W[f * NO + o];
  out[tid] = s;
}

// ---------------------------------------------------------------------------
extern "C" void kernel_launch(void* const* d_in, const int* in_sizes, int n_in,
                              void* d_out, int out_size, void* d_ws, size_t ws_size,
                              hipStream_t stream) {
  (void)in_sizes; (void)n_in; (void)out_size; (void)ws_size;
  const float* data    = (const float*)d_in[0];
  const float* mul_emb = (const float*)d_in[4];
  const float* phy_emb = (const float*)d_in[5];
  const float* net_emb = (const float*)d_in[6];
  const float* W_share = (const float*)d_in[7];
  const float* b_share = (const float*)d_in[8];
  const float* W_phy   = (const float*)d_in[9];
  const float* b_phy   = (const float*)d_in[10];
  const float* W_net   = (const float*)d_in[11];
  const float* b_net   = (const float*)d_in[12];
  const float* g_mul   = (const float*)d_in[13];
  const float* be_mul  = (const float*)d_in[14];
  const float* g_phy   = (const float*)d_in[15];
  const float* be_phy  = (const float*)d_in[16];
  const float* g_net   = (const float*)d_in[17];
  const float* be_net  = (const float*)d_in[18];
  const float* W_out   = (const float*)d_in[19];
  const float* b_out   = (const float*)d_in[20];
  const float* W_pout  = (const float*)d_in[21];
  const float* b_pout  = (const float*)d_in[22];
  const float* W_nout  = (const float*)d_in[23];
  const float* b_nout  = (const float*)d_in[24];

  const int N = 1000, Np = 1008, R = 32 * 1000;

  // Workspace layout (bytes, 16B-aligned). ~33 MB, fully L2-resident.
  char* ws = (char*)d_ws;
  float*    cosm  = (float*)   (ws + 0);         // 1000*1000*4 = 4,000,000
  _Float16* wn16  = (_Float16*)(ws + 4000000);   // 1008*64*2   =   129,024
  int*      idxm  = (int*)     (ws + 4160000);   // 1000*20*4   =    80,000
  int*      idxp  = (int*)     (ws + 4240000);
  int*      idxn  = (int*)     (ws + 4320000);
  float*    meanb = (float*)   (ws + 4400000);   // 64 floats
  float*    istdb = (float*)   (ws + 4400256);   // 64 floats
  _Float16* Bt16  = (_Float16*)(ws + 4400512);   // <= 64*64*2  =     8,192
  _Float16* A16   = (_Float16*)(ws + 4410000);   // 32000*64*2  = 4,096,000
  float*    bufA  = (float*)   (ws + 8510000);   // 32000*64*4  = 8,192,000
  float*    bufB  = (float*)   (ws + 16710000);
  float*    bufC  = (float*)   (ws + 24910000);  // ends ~33.1 MB

  // Output layout (all float32, tuple flattened in return order).
  float* out    = (float*)d_out;
  float* phy_o  = out;                // 32,000
  float* net_o  = out + 32000;        // 96,000
  float* idxm_f = out + 128000;       // 20,000
  float* x_net  = out + 148000;       // 2,048,000
  float* x_phy  = out + 2196000;      // 2,048,000

  auto gemm = [&](const _Float16* A, const _Float16* Bt, const float* bias,
                  float* C, int ldc, int M, int Kp, int Nc) {
    int Mp  = (M + 15) & ~15;
    int Ncp = (Nc + 15) & ~15;
    dim3 grid(Mp / 16, (Ncp + 63) / 64);
    wmma_gemm_packed_kernel<<<grid, 128, 0, stream>>>(A, Bt, bias, C, ldc,
                                                      M, Kp, Nc, Ncp);
  };
  auto packA = [&](const float* src, int lda, int M, int K, int Kp) {
    int Mp = (M + 15) & ~15;
    int total = Mp * Kp;
    pack_a_kernel<<<(total + 255) / 256, 256, 0, stream>>>(src, lda, M, K,
                                                           A16, Kp, total);
  };
  auto packBt = [&](const float* src, int ldb, int tB, int K, int Nc, int Kp) {
    int Ncp = (Nc + 15) & ~15;
    int total = Ncp * Kp;
    pack_bt_kernel<<<(total + 255) / 256, 256, 0, stream>>>(src, ldb, tB, K, Nc,
                                                            Bt16, Kp, total);
  };
  auto topk_graph = [&](const float* emb, int* idx_out, float* fidx_out) {
    normalize_pack_kernel<<<(Np + 255) / 256, 256, 0, stream>>>(emb, wn16, N, Np);
    gemm(wn16, wn16, nullptr, cosm, N, N, FDIM, N);  // cos = wn @ wn^T
    topk20_kernel<<<(N + 127) / 128, 128, 0, stream>>>(cosm, N, idx_out, fidx_out);
  };
  const int TOT = R * FDIM;
  auto bn = [&](const float* t, const float* g, const float* be, float* o) {
    colstats_kernel<<<FDIM, 256, 0, stream>>>(t, R, meanb, istdb);
    bn_apply_kernel<<<(TOT + 255) / 256, 256, 0, stream>>>(t, g, be, meanb,
                                                           istdb, o, TOT);
  };

  // ---- Stage A: mul graph -------------------------------------------------
  topk_graph(mul_emb, idxm, idxm_f);
  // h = data @ W_share + b_share          (32000 x 60) * (60 x 64), Kp=64
  packA(data, 60, R, 60, 64);
  packBt(W_share, FDIM, 0, 60, FDIM, 64);
  gemm(A16, Bt16, b_share, bufA, FDIM, R, 64, FDIM);
  // out = relu(mean gather) * mul_emb
  gather_mean_relu_kernel<<<(TOT + 255) / 256, 256, 0, stream>>>(
      bufA, idxm, mul_emb, bufB, N, TOT);
  bn(bufB, g_mul, be_mul, bufC);
  // out = bn @ W_out + b_out              (32000 x 64) * (64 x 64), Kp=64
  packA(bufC, FDIM, R, FDIM, 64);
  packBt(W_out, FDIM, 0, FDIM, FDIM, 64);
  gemm(A16, Bt16, b_out, bufA, FDIM, R, 64, FDIM);

  // ---- Stage B: phy branch ------------------------------------------------
  topk_graph(phy_emb, idxp, nullptr);
  // h_phy = out[..., :16] @ W_phy + b_phy (32000 x 16) * (16 x 64), Kp=32
  packA(bufA, FDIM, R, 16, 32);
  packBt(W_phy, FDIM, 0, 16, FDIM, 32);
  gemm(A16, Bt16, b_phy, bufB, FDIM, R, 32, FDIM);
  gather_mean_relu_kernel<<<(TOT + 255) / 256, 256, 0, stream>>>(
      bufB, idxp, nullptr, x_phy, N, TOT);
  rowmul_kernel<<<(TOT + 255) / 256, 256, 0, stream>>>(x_phy, phy_emb, bufC, N, TOT);
  bn(bufC, g_phy, be_phy, bufB);
  proj_kernel<<<(R * 1 + 255) / 256, 256, 0, stream>>>(bufB, W_pout, b_pout,
                                                       phy_o, R, 1);

  // ---- Stage C: net branch ------------------------------------------------
  topk_graph(net_emb, idxn, nullptr);
  // h_net = out[..., 16:] @ W_net + b_net (32000 x 48) * (48 x 64), Kp=64
  packA(bufA + 16, FDIM, R, 48, 64);
  packBt(W_net, FDIM, 0, 48, FDIM, 64);
  gemm(A16, Bt16, b_net, bufB, FDIM, R, 64, FDIM);
  gather_mean_relu_kernel<<<(TOT + 255) / 256, 256, 0, stream>>>(
      bufB, idxn, nullptr, x_net, N, TOT);
  rowmul_kernel<<<(TOT + 255) / 256, 256, 0, stream>>>(x_net, net_emb, bufC, N, TOT);
  bn(bufC, g_net, be_net, bufB);
  proj_kernel<<<(R * 3 + 255) / 256, 256, 0, stream>>>(bufB, W_nout, b_nout,
                                                       net_o, R, 3);
}
